// TCNNInstantNGP_3813930959287
// MI455X (gfx1250) — matrix operand order repaired and checked by reading, not compiled, not run on hardware
//
#include <hip/hip_runtime.h>
#include <math.h>

#define N_PTS         1048576
#define N_LEVELS      16
#define TBL           524288        // T = 1 << 19
#define PTS_PER_BLOCK 128
#define THREADS       256

// Padded LDS row strides (dwords) to kill bank conflicts:
// each is 16B-aligned and equals 4*odd dwords -> 16 rows hit 16 distinct bank groups.
#define FEAT_STRIDE   20            // dwords per point (16 used + 4 pad)
#define ACT_STRIDE    36            // dwords per row   (32 used + 4 pad)
#define W0T_STRIDE    20            // dwords per row   (16 used + 4 pad)  [32 bf16]
#define W1T_STRIDE    36            // dwords per row   (32 used + 4 pad)  [64 bf16]

typedef __bf16 v16bf __attribute__((ext_vector_type(16)));
typedef float  v8f   __attribute__((ext_vector_type(8)));

struct HGParams {
    float    scale[N_LEVELS];
    unsigned res[N_LEVELS];
    unsigned dense_mask;
};

// Pack two f32 into one dword of bf16 (lo in [15:0]), RNE.
// Single v_cvt_pk_bf16_f32 (present since gfx950; CDNA5 is bf16-centric).
__device__ __forceinline__ unsigned pack_bf16(float lo, float hi) {
    unsigned u;
    asm("v_cvt_pk_bf16_f32 %0, %1, %2" : "=v"(u) : "v"(lo), "v"(hi));
    return u;
}

// Lane exchange via ds_swizzle (group-of-32 xor mode): src lane = lane ^ MASK.
template <int OFF>
__device__ __forceinline__ float lane_swz(float v) {
    return __int_as_float(__builtin_amdgcn_ds_swizzle(__float_as_int(v), OFF));
}
#define SWZ_XOR(mask) (((mask) << 10) | 0x1F)

union Frag {
    uint4 q[2];
    v16bf v;
};

__global__ __launch_bounds__(THREADS)
void ngp_fused_kernel(const float* __restrict__ x,
                      const float* __restrict__ tables,
                      const float* __restrict__ w0,
                      const float* __restrict__ w1,
                      const float* __restrict__ w2,
                      float* __restrict__ out,
                      HGParams P) {
    // ~42 KB LDS total (WGP has 320 KB)
    __shared__ __align__(16) unsigned feats[PTS_PER_BLOCK * FEAT_STRIDE]; // [pt][16 bf16-pairs]
    __shared__ __align__(16) unsigned w0T[64 * W0T_STRIDE];               // [n][k-pairs] bf16 (transposed)
    __shared__ __align__(16) unsigned w1T[64 * W1T_STRIDE];               // [n][k-pairs] bf16 (transposed)
    __shared__ __align__(16) float    w2s[64];
    __shared__ __align__(16) unsigned actpk[8 * 16 * ACT_STRIDE];         // per-wave [16 rows][32 pairs]

    const int tid = threadIdx.x;

    // ------------- stage weights: fp32 global -> packed-bf16 LDS, transposed -------------
    // n in low tid bits -> consecutive lanes read consecutive floats (coalesced).
    #pragma unroll
    for (int i = 0; i < 4; ++i) {                 // w0T: 64 rows x 16 dwords = 1024
        int d  = tid + i * THREADS;
        int n  = d & 63, jp = d >> 6;             // row n, k-pair jp (0..15)
        w0T[n * W0T_STRIDE + jp] =
            pack_bf16(w0[(2 * jp) * 64 + n], w0[(2 * jp + 1) * 64 + n]);
    }
    #pragma unroll
    for (int i = 0; i < 8; ++i) {                 // w1T: 64 rows x 32 dwords = 2048
        int d  = tid + i * THREADS;
        int n  = d & 63, jp = d >> 6;             // jp (0..31)
        w1T[n * W1T_STRIDE + jp] =
            pack_bf16(w1[(2 * jp) * 64 + n], w1[(2 * jp + 1) * 64 + n]);
    }
    if (tid < 64) w2s[tid] = w2[tid];

    // ---------------- hash-grid encode: 2 threads/point, 8 levels each ----------------
    {
        const int pl   = tid >> 1;
        const int lvl0 = (tid & 1) * 8;
        const int gp   = blockIdx.x * PTS_PER_BLOCK + pl;
        const float x0 = (x[gp * 3 + 0] + 1.0f) * 0.5f;
        const float y0 = (x[gp * 3 + 1] + 1.0f) * 0.5f;
        const float z0 = (x[gp * 3 + 2] + 1.0f) * 0.5f;
        const float2* __restrict__ tab2 = (const float2*)tables;
        for (int li = 0; li < 8; ++li) {
            const int l   = lvl0 + li;
            const float s = P.scale[l];
            float px = x0 * s + 0.5f, py = y0 * s + 0.5f, pz = z0 * s + 0.5f;
            float gx = floorf(px), gy = floorf(py), gz = floorf(pz);
            float fx = px - gx, fy = py - gy, fz = pz - gz;
            unsigned cx = (unsigned)(int)gx, cy = (unsigned)(int)gy, cz = (unsigned)(int)gz;
            const unsigned r     = P.res[l];
            const bool     dense = (P.dense_mask >> l) & 1u;
            float f0 = 0.f, f1 = 0.f;
            #pragma unroll
            for (int c = 0; c < 8; ++c) {         // OFFS[c] = (c>>2, c>>1, c) & 1 (x,y,z)
                unsigned ox = (c >> 2) & 1u, oy = (c >> 1) & 1u, oz = c & 1u;
                unsigned ix = cx + ox, iy = cy + oy, iz = cz + oz;
                unsigned idx = dense
                    ? (ix + iy * r + iz * r * r)
                    : ((ix * 1u ^ iy * 2654435761u ^ iz * 805459861u) & (TBL - 1u));
                float wgt = (ox ? fx : 1.f - fx) * (oy ? fy : 1.f - fy) * (oz ? fz : 1.f - fz);
                float2 g = tab2[(size_t)l * TBL + idx];   // global_load_b64 gather (L2-resident)
                f0 = fmaf(wgt, g.x, f0);
                f1 = fmaf(wgt, g.y, f1);
            }
            feats[pl * FEAT_STRIDE + l] = pack_bf16(f0, f1); // one A-matrix K-pair per level
        }
    }

    __syncthreads();

    // ---------------- fused MLP: one 16-point tile per wave ----------------
    const int L   = tid & 31;
    const int wv  = tid >> 5;
    const int h16 = L >> 4;       // lane half: 0 -> K{0..7,16..23}, 1 -> K{8..15,24..31}
    const int m   = L & 15;
    const bool odd = (L & 1);

    // ---- layer 0: [16x32] @ [32x64] -> 4 WMMAs ----
    Frag a0;
    {
        const uint4* fp = (const uint4*)&feats[(wv * 16 + m) * FEAT_STRIDE];
        a0.q[0] = fp[h16];        // pairs {0..3} | {4..7}
        a0.q[1] = fp[h16 + 2];    // pairs {8..11} | {12..15}
    }
    v8f acc0[4];
    #pragma unroll
    for (int t = 0; t < 4; ++t) {
        Frag b;
        const uint4* wp = (const uint4*)&w0T[(t * 16 + m) * W0T_STRIDE];
        b.q[0] = wp[h16];         // K 0..7  | 8..15
        b.q[1] = wp[h16 + 2];     // K 16..23 | 24..31
        v8f z = {0.f, 0.f, 0.f, 0.f, 0.f, 0.f, 0.f, 0.f};
        acc0[t] = __builtin_amdgcn_wmma_f32_16x16x32_bf16(
            false, a0.v, false, b.v, (short)0, z, false, false);
    }
    #pragma unroll
    for (int t = 0; t < 4; ++t)
        #pragma unroll
        for (int r2 = 0; r2 < 8; ++r2)
            acc0[t][r2] = fmaxf(acc0[t][r2], 0.f);   // ReLU (dual-issues as v_dual_max)

    // ---- D-layout -> A-layout, divergence-free: both lanes of a column pair build the
    // same packed dword (parity-swapped lo/hi) and store it to the same LDS address.
    #pragma unroll
    for (int t = 0; t < 4; ++t) {
        #pragma unroll
        for (int r2 = 0; r2 < 8; ++r2) {
            float v  = acc0[t][r2];
            float o  = lane_swz<SWZ_XOR(1)>(v);      // neighbor column
            float lo = odd ? o : v;                  // even column -> low half
            float hi = odd ? v : o;                  // odd  column -> high half
            int row  = h16 * 8 + r2;                 // M = r | 8+r
            int pair = t * 8 + (m >> 1);             // column pair index
            actpk[(wv * 16 + row) * ACT_STRIDE + pair] = pack_bf16(lo, hi);
        }
    }

    // ---- layer 1: [16x64] @ [64x64] -> 8 WMMAs (2 K-slices x 4 N-tiles) ----
    v8f acc1[4];
    #pragma unroll
    for (int t = 0; t < 4; ++t)
        acc1[t] = (v8f){0.f, 0.f, 0.f, 0.f, 0.f, 0.f, 0.f, 0.f};
    #pragma unroll
    for (int s2 = 0; s2 < 2; ++s2) {
        Frag a;
        const uint4* ap = (const uint4*)&actpk[(wv * 16 + m) * ACT_STRIDE];
        a.q[0] = ap[s2 * 4 + h16];
        a.q[1] = ap[s2 * 4 + h16 + 2];
        #pragma unroll
        for (int t = 0; t < 4; ++t) {
            Frag b;
            const uint4* wp = (const uint4*)&w1T[(t * 16 + m) * W1T_STRIDE];
            b.q[0] = wp[s2 * 4 + h16];
            b.q[1] = wp[s2 * 4 + h16 + 2];
            acc1[t] = __builtin_amdgcn_wmma_f32_16x16x32_bf16(
                false, a.v, false, b.v, (short)0, acc1[t], false, false);
        }
    }
    #pragma unroll
    for (int t = 0; t < 4; ++t)
        #pragma unroll
        for (int r2 = 0; r2 < 8; ++r2)
            acc1[t][r2] = fmaxf(acc1[t][r2], 0.f);   // ReLU

    // ---- layer 2: [16x64] @ [64x1] via lane-dot + xor-swizzle tree ----
    float p[8] = {0.f, 0.f, 0.f, 0.f, 0.f, 0.f, 0.f, 0.f};
    #pragma unroll
    for (int t = 0; t < 4; ++t) {
        float w2v = w2s[t * 16 + m];
        #pragma unroll
        for (int r2 = 0; r2 < 8; ++r2)
            p[r2] = fmaf(acc1[t][r2], w2v, p[r2]);
    }
    #pragma unroll
    for (int r2 = 0; r2 < 8; ++r2) {          // reduce within each 16-lane half
        p[r2] += lane_swz<SWZ_XOR(1)>(p[r2]);
        p[r2] += lane_swz<SWZ_XOR(2)>(p[r2]);
        p[r2] += lane_swz<SWZ_XOR(4)>(p[r2]);
        p[r2] += lane_swz<SWZ_XOR(8)>(p[r2]);
    }

    if (m == 0) {                             // lanes 0 (M=0..7) and 16 (M=8..15)
        int g = blockIdx.x * PTS_PER_BLOCK + wv * 16 + h16 * 8;
        *(float4*)(out + g)     = make_float4(p[0], p[1], p[2], p[3]);
        *(float4*)(out + g + 4) = make_float4(p[4], p[5], p[6], p[7]);
    }
}

extern "C" void kernel_launch(void* const* d_in, const int* in_sizes, int n_in,
                              void* d_out, int out_size, void* d_ws, size_t ws_size,
                              hipStream_t stream) {
    (void)in_sizes; (void)n_in; (void)out_size; (void)d_ws; (void)ws_size;
    const float* x  = (const float*)d_in[0];
    const float* tb = (const float*)d_in[1];
    const float* w0 = (const float*)d_in[2];
    const float* w1 = (const float*)d_in[3];
    const float* w2 = (const float*)d_in[4];
    float* out = (float*)d_out;

    // Replicate the reference's float64 scale/res computation exactly on host.
    HGParams P;
    const double pls = pow(2048.0 / 16.0, 1.0 / 15.0);
    unsigned dm = 0;
    for (int l = 0; l < N_LEVELS; ++l) {
        double s   = 16.0 * pow(pls, (double)l) - 1.0;
        P.scale[l] = (float)s;
        unsigned r = (unsigned)ceil(s) + 1u;
        P.res[l]   = r;
        if ((unsigned long long)r * r * r <= (unsigned long long)TBL) dm |= (1u << l);
    }
    P.dense_mask = dm;

    dim3 grid(N_PTS / PTS_PER_BLOCK);
    ngp_fused_kernel<<<grid, THREADS, 0, stream>>>(x, tb, w0, w1, w2, out, P);
}
